// PairNN_240518169183
// MI455X (gfx1250) — matrix operand order repaired and checked by reading, not compiled, not run on hardware
//
#include <hip/hip_runtime.h>

#define NATOMS 5000
#define KNB    32
#define NPAIR  (NATOMS * KNB)
#define CCUT   3.0f
#define RMINC  3.5f
#define ETA    4.0f
#define NRAD   5
#define N3B    12
#define HID    128
#define DST    32          // padded desc stride in halves (17 real + 15 zero)
#define PI_F   3.14159265358979f

typedef __attribute__((ext_vector_type(16))) _Float16 v16h;
typedef __attribute__((ext_vector_type(8)))  float    v8f;

__device__ __forceinline__ float fast_rcp(float x) {
  return __builtin_amdgcn_rcpf(x);     // v_rcp_f32, ~1 ulp
}

// ---------------------------------------------------------------------------
// K1: per-pair geometry, cutoffs, radial Bessel basis (+ d/dr), desc cols 0..4
// ---------------------------------------------------------------------------
__global__ void k_geom(const float* __restrict__ rij,
                       float* __restrict__ u,  float* __restrict__ rr,
                       float* __restrict__ fc, float* __restrict__ dfc,
                       float* __restrict__ fk, float* __restrict__ dfk,
                       float* __restrict__ drbf,
                       _Float16* __restrict__ desc16) {
  int p = blockIdx.x * blockDim.x + threadIdx.x;
  if (p >= NPAIR) return;
  float x = rij[3*p+0], y = rij[3*p+1], z = rij[3*p+2];
  float r = sqrtf(x*x + y*y + z*z);
  float rinv = fast_rcp(fmaxf(r, 1e-12f));
  u[3*p+0] = x*rinv; u[3*p+1] = y*rinv; u[3*p+2] = z*rinv;
  rr[p] = r;
  float fcv = 1.f, dfcv = 0.f;
  if (r > RMINC) {
    float w = PI_F / (CCUT - RMINC);          // note: negative denominator
    float a = w * (r - RMINC);
    fcv  = 0.5f + 0.5f * __cosf(a);
    dfcv = -0.5f * w * __sinf(a);
  }
  fc[p] = fcv; dfc[p] = dfcv;
  float b = (PI_F / CCUT) * r;
  fk[p]  = 0.5f + 0.5f * __cosf(b);
  dfk[p] = -0.5f * (PI_F / CCUT) * __sinf(b);
  float pref = sqrtf(2.f / CCUT);
  float ir = rinv;
  #pragma unroll
  for (int i = 0; i < NRAD; ++i) {
    float kn = (float)(i + 1) * PI_F / CCUT;
    float sn = __sinf(kn * r), cn = __cosf(kn * r);
    float rbf = pref * sn * ir * fcv;
    float dr  = pref * ((kn * cn * r - sn) * ir * ir * fcv + sn * ir * dfcv);
    desc16[p*DST + i] = (_Float16)rbf;
    drbf[p*NRAD + i]  = dr;
  }
  #pragma unroll
  for (int k = NRAD + N3B; k < DST; ++k) desc16[p*DST + k] = (_Float16)0.f;
}

// ---------------------------------------------------------------------------
// K2: per-atom 3-body forward: dots(32x32, zero diag) -> g3b -> desc cols 5..16
// ---------------------------------------------------------------------------
__global__ void k_g3b(const float* __restrict__ u,
                      const float* __restrict__ fk,
                      _Float16* __restrict__ desc16) {
  __shared__ float su[KNB][3];
  __shared__ float sfk[KNB];
  __shared__ float sd[KNB][KNB];
  int n = blockIdx.x, t = threadIdx.x;
  int base = n * KNB;
  if (t < KNB) {
    su[t][0] = u[(base+t)*3+0];
    su[t][1] = u[(base+t)*3+1];
    su[t][2] = u[(base+t)*3+2];
    sfk[t]   = fk[base+t];
  }
  __syncthreads();
  for (int idx = t; idx < KNB*KNB; idx += blockDim.x) {
    int k = idx >> 5, l = idx & 31;
    float d = (k == l) ? 0.f
            : su[k][0]*su[l][0] + su[k][1]*su[l][1] + su[k][2]*su[l][2];
    sd[k][l] = d;
  }
  __syncthreads();
  for (int task = t; task < KNB * N3B; task += blockDim.x) {
    int k = task / N3B, m = task % N3B;
    float mu = -1.f + (2.f / 11.f) * (float)m;
    float acc = 0.f;
    #pragma unroll 4
    for (int l = 0; l < KNB; ++l) {
      float dd = sd[k][l] - mu;
      acc += __expf(-ETA * dd * dd) * sfk[l];
    }
    desc16[(base+k)*DST + NRAD + m] = (_Float16)acc;
  }
}

// ---------------------------------------------------------------------------
// K3: per-16-pair tile MLP fwd + bwd via WMMA (f16 in, f32 accumulate)
//   fwd:  Z(16x128) = desc(16x32pad) @ W1(32x16 per wave)  -- 1 wmma / wave
//   bwd:  gdesc(16x32pad) = G(16x128) @ W1^T               -- 4 wmma / wave(2)
// ---------------------------------------------------------------------------
__global__ void k_mlp(const _Float16* __restrict__ desc16,
                      const float* __restrict__ W1g,
                      const float* __restrict__ b1g,
                      const float* __restrict__ W2g,
                      const float* __restrict__ b2g,
                      const float* __restrict__ fcg,
                      float* __restrict__ gdesc,
                      float* __restrict__ outp) {
  __shared__ _Float16 sW1[DST][HID];   // rows >=17 zero
  __shared__ float    sb1[HID];
  __shared__ float    sW2[HID];
  __shared__ float    szb[16][HID];
  __shared__ _Float16 sG[16][HID];
  __shared__ _Float16 sA[16][DST];
  __shared__ float    sred[16][16];
  __shared__ float    sfc[16];
  int t = threadIdx.x;
  int pbase = blockIdx.x * 16;

  for (int i = t; i < DST * HID; i += 256) {
    int kk = i >> 7, nn = i & 127;
    sW1[kk][nn] = (kk < 17) ? (_Float16)W1g[kk*HID + nn] : (_Float16)0.f;
  }
  for (int i = t; i < HID; i += 256) { sb1[i] = b1g[i]; sW2[i] = W2g[i]; }
  for (int i = t; i < 16 * DST; i += 256) {
    int m = i >> 5, kk = i & 31;
    sA[m][kk] = desc16[(pbase+m)*DST + kk];
  }
  if (t < 16) sfc[t] = fcg[pbase + t];
  __syncthreads();

  int wave = t >> 5, lane = t & 31, lh = lane & 15;
  int hi = (lane >= 16) ? 1 : 0;

  // ---- forward: one 16x16x32 wmma per wave (wave w -> hidden cols w*16..+15)
  {
    v16h a, b;
    #pragma unroll
    for (int j = 0; j < 16; ++j) {
      int Ka = j + (hi ? 8 : 0) + ((j >= 8) ? 8 : 0);   // A 16x32 f16 layout
      a[j] = sA[lh][Ka];
      int Kb = j + (hi ? 8 : 0) + ((j >= 8) ? 8 : 0);   // B 32x16 f16 layout
      b[j] = sW1[Kb][wave*16 + lh];
    }
    v8f c = {0.f,0.f,0.f,0.f,0.f,0.f,0.f,0.f};
    c = __builtin_amdgcn_wmma_f32_16x16x32_f16(false, a, false, b,
                                               (short)0, c, false, false);
    #pragma unroll
    for (int i = 0; i < 8; ++i) {
      int M = i + (hi ? 8 : 0);
      int N = wave*16 + lh;
      szb[M][N] = c[i] + sb1[N];                          // z (pre-activation)
    }
  }
  __syncthreads();

  // ---- scalar: out = b2 + sum silu(z)*W2 ; G = fc*silu'(z)*W2 (f16)
  {
    int pg = t >> 4, j = t & 15;
    float acc = 0.f;
    for (int nn = j; nn < HID; nn += 16) {
      float zz = szb[pg][nn];
      float s  = fast_rcp(1.f + __expf(-zz));            // sigmoid, v_rcp_f32
      acc += zz * s * sW2[nn];
      float dsil = s * (1.f + zz * (1.f - s));
      sG[pg][nn] = (_Float16)(sfc[pg] * dsil * sW2[nn]);
    }
    sred[pg][j] = acc;
  }
  __syncthreads();
  if (t < 16) {
    float o = b2g[0];
    #pragma unroll
    for (int j = 0; j < 16; ++j) o += sred[t][j];
    outp[pbase + t] = o;
  }

  // ---- backward: gdesc = G @ W1^T ; waves 0,1 handle desc cols 0..15 / 16
  if (wave < 2) {
    v8f cg = {0.f,0.f,0.f,0.f,0.f,0.f,0.f,0.f};
    int nd = wave*16 + lh;                 // desc column for B (zero if >=17)
    for (int ks = 0; ks < 4; ++ks) {
      v16h a, b;
      #pragma unroll
      for (int j = 0; j < 16; ++j) {
        int Ka = j + (hi ? 8 : 0) + ((j >= 8) ? 8 : 0);
        a[j] = sG[lh][ks*32 + Ka];
        int Kb = ks*32 + j + (hi ? 8 : 0) + ((j >= 8) ? 8 : 0);
        b[j] = sW1[nd][Kb];                // B[k][n] = W1[nd][hidden k]
      }
      cg = __builtin_amdgcn_wmma_f32_16x16x32_f16(false, a, false, b,
                                                  (short)0, cg, false, false);
    }
    if (nd < 17) {
      #pragma unroll
      for (int i = 0; i < 8; ++i) {
        int M = i + (hi ? 8 : 0);
        gdesc[(pbase+M)*17 + nd] = cg[i];
      }
    }
  }
}

// ---------------------------------------------------------------------------
// K4: per-atom backward: E/DE matrices, column sums, B*u matvec, final beta
// ---------------------------------------------------------------------------
__global__ void k_back(const float* __restrict__ u,  const float* __restrict__ rr,
                       const float* __restrict__ dfc, const float* __restrict__ fk,
                       const float* __restrict__ dfk, const float* __restrict__ drbf,
                       const float* __restrict__ gdesc, const float* __restrict__ outp,
                       float* __restrict__ beta) {
  __shared__ float su[KNB][3];
  __shared__ float sr[KNB], sfk[KNB], sdfk[KNB], scoef[KNB], sC[KNB];
  __shared__ float ss[KNB][N3B];
  __shared__ float sE[KNB][KNB];
  __shared__ float sDE[KNB][KNB];
  int n = blockIdx.x, t = threadIdx.x;
  int base = n * KNB;
  if (t < KNB) {
    su[t][0] = u[(base+t)*3+0];
    su[t][1] = u[(base+t)*3+1];
    su[t][2] = u[(base+t)*3+2];
    sr[t]   = rr[base+t];
    sfk[t]  = fk[base+t];
    sdfk[t] = dfk[base+t];
    float cr = outp[base+t] * dfc[base+t];
    #pragma unroll
    for (int i = 0; i < NRAD; ++i)
      cr += gdesc[(base+t)*17 + i] * drbf[(base+t)*NRAD + i];
    scoef[t] = cr;
  }
  for (int i = t; i < KNB * N3B; i += blockDim.x) {
    int k = i / N3B, m = i % N3B;
    ss[k][m] = gdesc[(base+k)*17 + NRAD + m];
  }
  __syncthreads();
  for (int idx = t; idx < KNB*KNB; idx += blockDim.x) {
    int k = idx >> 5, l = idx & 31;
    float dd = (k == l) ? 0.f
             : su[k][0]*su[l][0] + su[k][1]*su[l][1] + su[k][2]*su[l][2];
    float es = 0.f, ds = 0.f;
    #pragma unroll
    for (int m = 0; m < N3B; ++m) {
      float mu = -1.f + (2.f / 11.f) * (float)m;
      float tt = dd - mu;
      float e  = __expf(-ETA * tt * tt);
      float sm = ss[k][m];
      es += sm * e;
      ds += sm * e * tt;
    }
    sE[k][l]  = es;                                  // keep diagonal (fcrik path)
    sDE[k][l] = (k == l) ? 0.f : (-2.f * ETA) * ds;  // zero diag (fill_diagonal)
  }
  __syncthreads();
  if (t < KNB) {                                     // C_l = column sums of E
    float c = 0.f;
    #pragma unroll 4
    for (int k = 0; k < KNB; ++k) c += sE[k][t];
    sC[t] = c;
  }
  __syncthreads();
  if (t < KNB) {
    int k = t;
    float tx = 0.f, ty = 0.f, tz = 0.f;
    #pragma unroll 4
    for (int l = 0; l < KNB; ++l) {
      if (l == k) continue;
      float w = sDE[k][l] * sfk[l] + sDE[l][k] * sfk[k];
      tx += w * su[l][0]; ty += w * su[l][1]; tz += w * su[l][2];
    }
    float ux = su[k][0], uy = su[k][1], uz = su[k][2];
    float udt = ux*tx + uy*ty + uz*tz;
    float rin = fast_rcp(sr[k]);
    float rad = scoef[k] + sC[k] * sdfk[k];
    beta[(base+k)*3+0] = (tx - udt*ux) * rin + rad * ux;
    beta[(base+k)*3+1] = (ty - udt*uy) * rin + rad * uy;
    beta[(base+k)*3+2] = (tz - udt*uz) * rin + rad * uz;
  }
}

// ---------------------------------------------------------------------------
extern "C" void kernel_launch(void* const* d_in, const int* in_sizes, int n_in,
                              void* d_out, int out_size, void* d_ws, size_t ws_size,
                              hipStream_t stream) {
  const float* rij = (const float*)d_in[0];
  // d_in[1] = unique_i (int32) -- pairs are grouped K-per-atom; unused
  const float* W1  = (const float*)d_in[2];
  const float* b1  = (const float*)d_in[3];
  const float* W2  = (const float*)d_in[4];
  const float* b2  = (const float*)d_in[5];
  float* beta = (float*)d_out;

  const size_t Pn = (size_t)NPAIR;
  float* f = (float*)d_ws;
  float* u      = f;                 // 3P
  float* rr     = f + 3*Pn;          // P
  float* fc     = f + 4*Pn;          // P
  float* dfc    = f + 5*Pn;          // P
  float* fk     = f + 6*Pn;          // P
  float* dfk    = f + 7*Pn;          // P
  float* drbf   = f + 8*Pn;          // 5P
  float* gdesc  = f + 13*Pn;         // 17P
  float* outp   = f + 30*Pn;         // P
  _Float16* desc16 = (_Float16*)(f + 31*Pn);  // 32P halves

  k_geom<<<(NPAIR + 255) / 256, 256, 0, stream>>>(rij, u, rr, fc, dfc, fk, dfk,
                                                  drbf, desc16);
  k_g3b<<<NATOMS, 256, 0, stream>>>(u, fk, desc16);
  k_mlp<<<NPAIR / 16, 256, 0, stream>>>(desc16, W1, b1, W2, b2, fc, gdesc, outp);
  k_back<<<NATOMS, 256, 0, stream>>>(u, rr, dfc, fk, dfk, drbf, gdesc, outp, beta);
}
